// TKL_MF_42674795053915
// MI455X (gfx1250) — matrix operand order repaired and checked by reading, not compiled, not run on hardware
//
#include <hip/hip_runtime.h>
#include <hip/hip_bf16.h>
#include <math.h>

typedef __attribute__((ext_vector_type(16))) _Float16 v16h;
typedef __attribute__((ext_vector_type(8)))  float    v8f;

// ---------------- model dims ----------------
#define BQ_   16
#define QL_   30
#define DL_   2000
#define D_    768
#define H_    8
#define HD_   96
#define FF_   3072
#define OV_   7
#define SQ_   30      // query seq len
#define SD_   65      // doc chunk seq len (EXT+1)
#define TQ_   (BQ_*SQ_)   // 480 tokens
#define TD_   (BQ_*SD_)   // 1040 tokens

#define KCH_STEPS 16      // K-steps (x32) staged in LDS per chunk = 512 cols

// ---------------- helpers ----------------
__device__ __forceinline__ float gelu_tanh(float x) {
    const float c = 0.7978845608028654f;
    float t = tanhf(c * (x + 0.044715f * x * x * x));
    return 0.5f * x * (1.0f + t);
}

// ---------------- weight repack: f32 [K,N] row-major -> f16 WMMA B-fragments ---------
// Packed layout: p = ((ntile*ksteps + kk)*32 + lane)*16 + e
//   lane<16 : N = ntile*16+lane,    K = kk*32 + e        (e = 0..15)
//   lane>=16: N = ntile*16+lane-16, K = kk*32 + 16 + e
__global__ __launch_bounds__(256) void TKL_pack_w(const float* __restrict__ W,
                                                  _Float16* __restrict__ dst,
                                                  int K, int N) {
    const int ksteps = K >> 5;
    const long total = (long)K * (long)N;
    for (long p = (long)blockIdx.x * 256 + threadIdx.x; p < total;
         p += (long)gridDim.x * 256) {
        int  e    = (int)(p & 15);
        int  lane = (int)((p >> 4) & 31);
        long q    = p >> 9;
        int  kk   = (int)(q % ksteps);
        int  nt   = (int)(q / ksteps);
        int  n    = nt * 16 + (lane & 15);
        int  k    = (kk << 5) + ((lane & 16) ? 16 : 0) + e;
        dst[p] = (_Float16)W[(long)k * N + n];
    }
}

// ---------------- WMMA GEMM: C[M,N] = A[M,K](f32) x Bpacked(f16) + bias, optional GELU
// 256 threads = 8 waves; all waves share one 16-row A tile staged in LDS in
// A-fragment order; wave w computes N-tile (blockIdx.y*8 + w).
__global__ __launch_bounds__(256) void TKL_gemm_wmma(const float* __restrict__ A,
                                                     const _Float16* __restrict__ Bp,
                                                     const float* __restrict__ bias,
                                                     float* __restrict__ C,
                                                     int M, int N, int K, int epi) {
    __shared__ _Float16 sA[KCH_STEPS * 512];   // 16 KB
    const int tid    = threadIdx.x;
    const int wave   = tid >> 5;
    const int lane   = tid & 31;
    const int m0     = blockIdx.x * 16;
    const int ksteps = K >> 5;
    const int ntiles = N >> 4;
    const int ntile  = blockIdx.y * 8 + wave;

    v8f acc = {};
    const _Float16* bptr = Bp + (long)ntile * ksteps * 512;

    for (int kb = 0; kb < ksteps; kb += KCH_STEPS) {
        int nst = ksteps - kb;
        if (nst > KCH_STEPS) nst = KCH_STEPS;
        __syncthreads();
        // stage A fragments: A-matrix 16x32 f16 layout (contiguous k, matches packed B):
        //   lane<16 : K = k0 + e
        //   lane>=16: K = k0 + 16 + e
        const int nelem = nst * 512;
        for (int p = tid; p < nelem; p += 256) {
            int e   = p & 15;
            int l   = (p >> 4) & 31;
            int kkl = p >> 9;
            int row = m0 + (l & 15);
            int kc  = ((kb + kkl) << 5) + e + ((l & 16) ? 16 : 0);
            sA[p] = (_Float16)A[(long)row * K + kc];
        }
        __syncthreads();
        if (ntile < ntiles) {
            for (int kk = 0; kk < nst; ++kk) {
                v16h a = *(const v16h*)(sA + kk * 512 + lane * 16);
                v16h b = *(const v16h*)(bptr + (long)(kb + kk) * 512 + lane * 16);
                acc = __builtin_amdgcn_wmma_f32_16x16x32_f16(
                        false, a, false, b, (short)0, acc, false, false);
            }
        }
    }

    if (ntile < ntiles) {
        const int n     = ntile * 16 + (lane & 15);
        const float bv  = bias ? bias[n] : 0.0f;
        const int mbase = m0 + ((lane & 16) ? 8 : 0);
        #pragma unroll
        for (int r = 0; r < 8; ++r) {
            float v = acc[r] + bv;
            if (epi == 1) v = gelu_tanh(v);
            C[(long)(mbase + r) * N + n] = v;
        }
    }
}

// ---------------- attention: one block per (sequence, head) ----------------
__global__ __launch_bounds__(256) void TKL_attn(const float* __restrict__ QKV,
                                                float* __restrict__ O, int S) {
    __shared__ float    sQ[SD_ * HD_];   // reused for V (24.96 KB)
    __shared__ _Float16 sK[SD_ * HD_];   // 12.48 KB
    __shared__ float    sS[SD_ * 66];    // 17.16 KB  (stride 66 vs banks)
    const int tid  = threadIdx.x;
    const int seq  = blockIdx.x >> 3;
    const int h    = blockIdx.x & 7;
    const int qoff = h * HD_;
    const int koff = D_ + h * HD_;
    const int voff = 2 * D_ + h * HD_;

    for (int p = tid; p < S * HD_; p += 256) {
        int t = p / HD_, d = p - t * HD_;
        const float* row = QKV + (long)(seq * S + t) * (3 * D_);
        sQ[p] = row[qoff + d];
        sK[p] = (_Float16)row[koff + d];
    }
    __syncthreads();

    const float scale = 0.10206207262f;   // 1/sqrt(96)
    for (int p = tid; p < S * S; p += 256) {
        int q = p / S, k = p - q * S;
        float a = 0.0f;
        #pragma unroll 8
        for (int d = 0; d < HD_; ++d) a += sQ[q * HD_ + d] * (float)sK[k * HD_ + d];
        sS[q * 66 + k] = a * scale;
    }
    __syncthreads();

    for (int q = tid; q < S; q += 256) {
        float mx = -1e30f;
        for (int k = 0; k < S; ++k) mx = fmaxf(mx, sS[q * 66 + k]);
        float sum = 0.0f;
        for (int k = 0; k < S; ++k) { float e = __expf(sS[q * 66 + k] - mx); sS[q * 66 + k] = e; sum += e; }
        float inv = 1.0f / sum;
        for (int k = 0; k < S; ++k) sS[q * 66 + k] *= inv;
    }
    __syncthreads();

    for (int p = tid; p < S * HD_; p += 256) {   // V into sQ
        int t = p / HD_, d = p - t * HD_;
        sQ[p] = QKV[(long)(seq * S + t) * (3 * D_) + voff + d];
    }
    __syncthreads();

    for (int p = tid; p < S * HD_; p += 256) {
        int q = p / HD_, d = p - q * HD_;
        float a = 0.0f;
        for (int k = 0; k < S; ++k) a += sS[q * 66 + k] * sQ[k * HD_ + d];
        O[(long)(seq * S + q) * D_ + h * HD_ + d] = a;
    }
}

// ---------------- residual + layernorm (in place on X) ----------------
__global__ __launch_bounds__(256) void TKL_res_ln(float* __restrict__ X,
                                                  const float* __restrict__ Y,
                                                  const float* __restrict__ g,
                                                  const float* __restrict__ b) {
    __shared__ float rs[256], rq[256];
    const int row = blockIdx.x, tid = threadIdx.x;
    float v[3];
    float s = 0.0f, sq = 0.0f;
    #pragma unroll
    for (int j = 0; j < 3; ++j) {
        int i = tid + j * 256;
        float t = X[(long)row * D_ + i] + Y[(long)row * D_ + i];
        v[j] = t; s += t; sq += t * t;
    }
    rs[tid] = s; rq[tid] = sq; __syncthreads();
    for (int st = 128; st > 0; st >>= 1) {
        if (tid < st) { rs[tid] += rs[tid + st]; rq[tid] += rq[tid + st]; }
        __syncthreads();
    }
    const float mean = rs[0] * (1.0f / D_);
    const float var  = rq[0] * (1.0f / D_) - mean * mean;
    const float rn   = rsqrtf(var + 1e-5f);
    #pragma unroll
    for (int j = 0; j < 3; ++j) {
        int i = tid + j * 256;
        X[(long)row * D_ + i] = (v[j] - mean) * rn * g[i] + b[i];
    }
}

// ---------------- input builders ----------------
__global__ __launch_bounds__(256) void TKL_build_q(const float* __restrict__ emb,
                                                   const float* __restrict__ pos_q,
                                                   const int* __restrict__ query,
                                                   float* __restrict__ X,
                                                   float* __restrict__ X0) {
    const int r = blockIdx.x, tid = threadIdx.x;
    const int b = r / SQ_, s = r - b * SQ_;
    const int tok = query[b * QL_ + s];
    for (int i = tid; i < D_; i += 256) {
        float v = emb[(long)tok * D_ + i] + pos_q[(long)s * D_ + i];
        X[(long)r * D_ + i] = v; X0[(long)r * D_ + i] = v;
    }
}

// doc chunk 0 only: token0 = cls_topic[0]; token t>=1 -> d_pad col (t-1):
// zeros for t-1 < OV, else emb[document[b, t-1-OV]] + pos_d[t-1-OV]
__global__ __launch_bounds__(256) void TKL_build_d(const float* __restrict__ emb,
                                                   const float* __restrict__ pos_d,
                                                   const float* __restrict__ cls0,
                                                   const int* __restrict__ doc,
                                                   float* __restrict__ X,
                                                   float* __restrict__ X0) {
    const int r = blockIdx.x, tid = threadIdx.x;
    const int b = r / SD_, t = r - b * SD_;
    float* xr  = X  + (long)r * D_;
    float* x0r = X0 + (long)r * D_;
    if (t == 0) {
        for (int i = tid; i < D_; i += 256) { float v = cls0[i]; xr[i] = v; x0r[i] = v; }
    } else {
        const int j = t - 1;
        if (j >= OV_) {
            const int di  = j - OV_;
            const int tok = doc[b * DL_ + di];
            for (int i = tid; i < D_; i += 256) {
                float v = emb[(long)tok * D_ + i] + pos_d[(long)di * D_ + i];
                xr[i] = v; x0r[i] = v;
            }
        } else {
            for (int i = tid; i < D_; i += 256) { xr[i] = 0.0f; x0r[i] = 0.0f; }
        }
    }
}

// ---------------- mix + pool (x @ W.T + b) + L2 normalize ----------------
__global__ __launch_bounds__(256) void TKL_pool_norm(const float* __restrict__ Xfin,
                                                     const float* __restrict__ X0,
                                                     const float* __restrict__ mixer,
                                                     const float* __restrict__ W,
                                                     const float* __restrict__ bias,
                                                     float* __restrict__ out,
                                                     int S) {
    __shared__ float ctx[D_];
    __shared__ float vv[D_];
    __shared__ float red[256];
    const int b = blockIdx.x, tid = threadIdx.x;
    const float mx = mixer[0];
    const long r = (long)b * S;   // CLS token row
    for (int i = tid; i < D_; i += 256)
        ctx[i] = mx * X0[r * D_ + i] + (1.0f - mx) * Xfin[r * D_ + i];
    __syncthreads();
    for (int n = tid; n < D_; n += 256) {
        float a = bias[n];
        const float* wr = W + (long)n * D_;   // out = x @ W.T  -> row n of W
        for (int d = 0; d < D_; ++d) a += ctx[d] * wr[d];
        vv[n] = a;
    }
    __syncthreads();
    float s = 0.0f;
    for (int n = tid; n < D_; n += 256) s += vv[n] * vv[n];
    red[tid] = s; __syncthreads();
    for (int st = 128; st > 0; st >>= 1) { if (tid < st) red[tid] += red[tid + st]; __syncthreads(); }
    const float inv = 1.0f / (sqrtf(red[0]) + 1e-4f);
    for (int n = tid; n < D_; n += 256) out[(long)b * D_ + n] = vv[n] * inv;
}

__global__ __launch_bounds__(256) void TKL_score(float* __restrict__ score,
                                                 const float* __restrict__ qv,
                                                 const float* __restrict__ dv) {
    __shared__ float red[256];
    const int b = blockIdx.x, tid = threadIdx.x;
    float s = 0.0f;
    for (int i = tid; i < D_; i += 256) s += qv[(long)b * D_ + i] * dv[(long)b * D_ + i];
    red[tid] = s; __syncthreads();
    for (int st = 128; st > 0; st >>= 1) { if (tid < st) red[tid] += red[tid + st]; __syncthreads(); }
    if (tid == 0) score[b] = red[0] * 6.0f;
}

// ---------------- host orchestration ----------------
static void run_encoder(float* X, float* QKV, float* ATTO, float* FF1, float* Y,
                        const _Float16* pqkv, const _Float16* po,
                        const _Float16* pw1, const _Float16* pw2,
                        const float* bqkv, const float* bo,
                        const float* g1, const float* b1,
                        const float* bf1, const float* bf2,
                        const float* g2, const float* b2,
                        int T, int nseq, int S, hipStream_t stream) {
    const long H_QKV = 768L * 2304, H_O = 768L * 768, H_W1 = 768L * 3072, H_W2 = 3072L * 768;
    for (int l = 0; l < 2; ++l) {
        TKL_gemm_wmma<<<dim3(T / 16, 2304 / 128), 256, 0, stream>>>(
            X, pqkv + l * H_QKV, bqkv + l * 2304, QKV, T, 2304, 768, 0);
        TKL_attn<<<nseq * H_, 256, 0, stream>>>(QKV, ATTO, S);
        TKL_gemm_wmma<<<dim3(T / 16, 768 / 128), 256, 0, stream>>>(
            ATTO, po + l * H_O, bo + l * 768, Y, T, 768, 768, 0);
        TKL_res_ln<<<T, 256, 0, stream>>>(X, Y, g1 + l * 768, b1 + l * 768);
        TKL_gemm_wmma<<<dim3(T / 16, 3072 / 128), 256, 0, stream>>>(
            X, pw1 + l * H_W1, bf1 + l * 3072, FF1, T, 3072, 768, 1);
        TKL_gemm_wmma<<<dim3(T / 16, 768 / 128), 256, 0, stream>>>(
            FF1, pw2 + l * H_W2, bf2 + l * 768, Y, T, 768, 3072, 0);
        TKL_res_ln<<<T, 256, 0, stream>>>(X, Y, g2 + l * 768, b2 + l * 768);
    }
}

extern "C" void kernel_launch(void* const* d_in, const int* in_sizes, int n_in,
                              void* d_out, int out_size, void* d_ws, size_t ws_size,
                              hipStream_t stream) {
    (void)in_sizes; (void)n_in; (void)out_size; (void)ws_size;
    const float* emb    = (const float*)d_in[0];
    const float* pos_q  = (const float*)d_in[1];
    const float* pos_d  = (const float*)d_in[2];
    const float* cls_t  = (const float*)d_in[3];
    const float* mixer  = (const float*)d_in[4];
    const float* poolW  = (const float*)d_in[5];
    const float* poolB  = (const float*)d_in[6];
    const float* Wqkv   = (const float*)d_in[7];
    const float* bqkv   = (const float*)d_in[8];
    const float* Wo     = (const float*)d_in[9];
    const float* bo     = (const float*)d_in[10];
    const float* g1     = (const float*)d_in[11];
    const float* b1     = (const float*)d_in[12];
    const float* W1     = (const float*)d_in[13];
    const float* bf1    = (const float*)d_in[14];
    const float* W2     = (const float*)d_in[15];
    const float* bf2    = (const float*)d_in[16];
    const float* g2     = (const float*)d_in[17];
    const float* b2     = (const float*)d_in[18];
    const int*   query  = (const int*)d_in[19];
    const int*   doc    = (const int*)d_in[20];
    float* out = (float*)d_out;

    // ---- workspace carve-up ----
    const long H_QKV = 768L * 2304, H_O = 768L * 768, H_W1 = 768L * 3072, H_W2 = 3072L * 768;
    _Float16* hb     = (_Float16*)d_ws;
    _Float16* pqkv   = hb;                          // 2 * H_QKV
    _Float16* po     = pqkv + 2 * H_QKV;            // 2 * H_O
    _Float16* pw1    = po   + 2 * H_O;              // 2 * H_W1
    _Float16* pw2    = pw1  + 2 * H_W1;             // 2 * H_W2
    float* fb   = (float*)(pw2 + 2 * H_W2);         // byte off 28,311,552 (256B aligned)
    float* Xq   = fb;                               // 480*768
    float* X0q  = Xq  + (long)TQ_ * D_;
    float* Xd   = X0q + (long)TQ_ * D_;             // 1040*768
    float* X0d  = Xd  + (long)TD_ * D_;
    float* QKV  = X0d + (long)TD_ * D_;             // 1040*2304
    float* ATTO = QKV + (long)TD_ * 3 * D_;         // 1040*768
    float* FF1  = ATTO + (long)TD_ * D_;            // 1040*3072
    float* Y    = FF1 + (long)TD_ * FF_;            // 1040*768

    // ---- pack all weights to WMMA-fragment f16 ----
    for (int l = 0; l < 2; ++l) {
        TKL_pack_w<<<1024, 256, 0, stream>>>(Wqkv + (long)l * 768 * 2304, pqkv + l * H_QKV, 768, 2304);
        TKL_pack_w<<<1024, 256, 0, stream>>>(Wo   + (long)l * 768 * 768,  po   + l * H_O,   768, 768);
        TKL_pack_w<<<1024, 256, 0, stream>>>(W1   + (long)l * 768 * 3072, pw1  + l * H_W1,  768, 3072);
        TKL_pack_w<<<1024, 256, 0, stream>>>(W2   + (long)l * 3072 * 768, pw2  + l * H_W2,  3072, 768);
    }

    // ---- build inputs ----
    TKL_build_q<<<TQ_, 256, 0, stream>>>(emb, pos_q, query, Xq, X0q);
    TKL_build_d<<<TD_, 256, 0, stream>>>(emb, pos_d, cls_t /*row 0*/, doc, Xd, X0d);

    // ---- encoders (query: 16 seqs x30 ; doc: only chunk 0 matters -> 16 seqs x65) ----
    run_encoder(Xq, QKV, ATTO, FF1, Y, pqkv, po, pw1, pw2,
                bqkv, bo, g1, b1, bf1, bf2, g2, b2, TQ_, BQ_, SQ_, stream);
    run_encoder(Xd, QKV, ATTO, FF1, Y, pqkv, po, pw1, pw2,
                bqkv, bo, g1, b1, bf1, bf2, g2, b2, TD_, BQ_, SD_, stream);

    // ---- pool, normalize, score ----
    float* qv = out + BQ_;
    float* dv = out + BQ_ + (long)BQ_ * D_;
    TKL_pool_norm<<<BQ_, 256, 0, stream>>>(Xq, X0q, mixer, poolW, poolB, qv, SQ_);
    TKL_pool_norm<<<BQ_, 256, 0, stream>>>(Xd, X0d, mixer, poolW, poolB, dv, SD_);
    TKL_score<<<BQ_, 256, 0, stream>>>(out, qv, dv);
}
